// QuantizedLinear_40432822125163
// MI455X (gfx1250) — matrix operand order repaired and checked by reading, not compiled, not run on hardware
//
#include <hip/hip_runtime.h>

typedef __attribute__((ext_vector_type(16))) __bf16 v16bf;
typedef __attribute__((ext_vector_type(8)))  __bf16 v8bf;
typedef __attribute__((ext_vector_type(4)))  __bf16 v4bf;
typedef __attribute__((ext_vector_type(8)))  float  v8f;
typedef __attribute__((ext_vector_type(4)))  float  f4;
typedef __attribute__((ext_vector_type(4)))  int    i4;

#define TOKENS_N 8192
#define IN_F     4096
#define OUT_F    11008

constexpr int BM  = 128;          // output rows per block
constexpr int BN  = 128;          // output cols per block
constexpr int BK  = 32;           // K step (matches wmma 16x16x32)
constexpr int LDT = BK + 8;       // padded LDS row stride (bf16) -> conflict-free b128
constexpr int NK  = IN_F / BK;    // 128 K-steps

// ---- fp32 -> (hi,lo) bf16 split, round-to-nearest-even ----------------------
__device__ __forceinline__ unsigned short f2bf_rne(float f) {
  unsigned u = __builtin_bit_cast(unsigned, f);
  unsigned r = u + 0x7FFFu + ((u >> 16) & 1u);
  return (unsigned short)(r >> 16);
}
__device__ __forceinline__ void split_bf16(float f, __bf16& hi, __bf16& lo) {
  unsigned short hb = f2bf_rne(f);
  hi = __builtin_bit_cast(__bf16, hb);
  float hf = __builtin_bit_cast(float, (unsigned)hb << 16);
  lo = __builtin_bit_cast(__bf16, f2bf_rne(f - hf));
}

// ---- WMMA wrapper -----------------------------------------------------------
__device__ __forceinline__ v8f wmma_bf16(v16bf a, v16bf b, v8f c) {
  // (neg_a, A, neg_b, B, c_mod, C, reuse_a, reuse_b)
  return __builtin_amdgcn_wmma_f32_16x16x32_bf16(false, a, false, b, (short)0, c,
                                                 false, false);
}

// ---- fragment loads per ISA 7.12.2 layouts ---------------------------------
// A (16x32 MxK, 16-bit): lanes 0-15 hold M=lane, K = {0..7, 16..23};
//                        lanes 16-31 hold M=lane-16, K = {8..15, 24..31}.
__device__ __forceinline__ v16bf load_fragA(const __bf16* s, int r0, int lane) {
  int r = r0 + (lane & 15);
  int k = (lane >> 4) << 3;                 // 0 or 8
  const __bf16* p = s + r * LDT + k;
  v8bf x = *(const v8bf*)p;                 // K = k .. k+7
  v8bf y = *(const v8bf*)(p + 16);          // K = k+16 .. k+23
  v16bf o;
#pragma unroll
  for (int i = 0; i < 8; ++i) { o[i] = x[i]; o[i + 8] = y[i]; }
  return o;
}
// B (32x16 KxN, 16-bit): lanes 0-15 hold N=lane, K = {0..7, 16..23};
//                        lanes 16-31 hold N=lane-16, K = {8..15, 24..31}.
// Our B[k][n] = W[n][k]; LDS holds W-tile row-major [n][k] -> contiguous reads.
__device__ __forceinline__ v16bf load_fragB(const __bf16* s, int n0, int lane) {
  int r = n0 + (lane & 15);
  int k = (lane >> 4) << 3;                 // 0 or 8
  const __bf16* p = s + r * LDT + k;
  v8bf x = *(const v8bf*)p;
  v8bf y = *(const v8bf*)(p + 16);
  v16bf o;
#pragma unroll
  for (int i = 0; i < 8; ++i) { o[i] = x[i]; o[i + 8] = y[i]; }
  return o;
}

__global__ __launch_bounds__(256)
void bitnet_wmma_gemm(const float* __restrict__ A, const int* __restrict__ W,
                      const float* __restrict__ scale, const float* __restrict__ bias,
                      float* __restrict__ out) {
  __shared__ __align__(16) __bf16 sAh[2][BM * LDT];  // A hi tile
  __shared__ __align__(16) __bf16 sAl[2][BM * LDT];  // A lo tile (residual)
  __shared__ __align__(16) __bf16 sW [2][BM * LDT];  // W tile (bf16 of {0,1,2})

  const int t     = threadIdx.x;
  const int lane  = t & 31;
  const int warp  = t >> 5;            // 8 waves
  const int waveM = (warp & 1) * 64;   // 2 waves along M (64 rows each)
  const int waveN = (warp >> 1) * 32;  // 4 waves along N (32 cols each)

  // Cooperative loader mapping: flat float4 index f = t + 256*i,
  // row = f>>3 (0..127), col = (f&7)*4  -> fully coalesced 16B per lane.
  const int rowB = t >> 3;             // 0..31, +32*i
  const int col  = (t & 7) << 2;       // 0,4,..,28

  const float* gA = A + (size_t)(blockIdx.x * BM + rowB) * IN_F + col;
  const int*   gW = W + (size_t)(blockIdx.y * BN + rowB) * IN_F + col;

  f4 ar[4];
  i4 wr[4];

  auto loadG = [&](int kt) {
#pragma unroll
    for (int i = 0; i < 4; ++i) {
      ar[i] = *(const f4*)(gA + (size_t)i * 32 * IN_F + kt * BK);
      wr[i] = *(const i4*)(gW + (size_t)i * 32 * IN_F + kt * BK);
    }
  };
  auto storeLds = [&](int buf) {
#pragma unroll
    for (int i = 0; i < 4; ++i) {
      int r = rowB + i * 32;
      v4bf h, l, w;
#pragma unroll
      for (int j = 0; j < 4; ++j) {
        __bf16 hb, lb;
        split_bf16(ar[i][j], hb, lb);
        h[j] = hb; l[j] = lb;
        w[j] = __builtin_bit_cast(__bf16, f2bf_rne((float)wr[i][j]));
      }
      *(v4bf*)&sAh[buf][r * LDT + col] = h;
      *(v4bf*)&sAl[buf][r * LDT + col] = l;
      *(v4bf*)&sW [buf][r * LDT + col] = w;
    }
  };

  v8f acc[4][2];
#pragma unroll
  for (int m = 0; m < 4; ++m)
#pragma unroll
    for (int n = 0; n < 2; ++n) acc[m][n] = v8f{};

  loadG(0);
  storeLds(0);
  __syncthreads();

  for (int kt = 0; kt < NK; ++kt) {
    const int cur = kt & 1;
    const bool more = (kt + 1) < NK;
    if (more) loadG(kt + 1);           // global prefetch into registers

    v16bf b0 = load_fragB(sW[cur], waveN + 0,  lane);
    v16bf b1 = load_fragB(sW[cur], waveN + 16, lane);
#pragma unroll
    for (int mt = 0; mt < 4; ++mt) {
      v16bf ah = load_fragA(sAh[cur], waveM + mt * 16, lane);
      v16bf al = load_fragA(sAl[cur], waveM + mt * 16, lane);
      acc[mt][0] = wmma_bf16(ah, b0, acc[mt][0]);
      acc[mt][0] = wmma_bf16(al, b0, acc[mt][0]);
      acc[mt][1] = wmma_bf16(ah, b1, acc[mt][1]);
      acc[mt][1] = wmma_bf16(al, b1, acc[mt][1]);
    }

    if (more) storeLds(cur ^ 1);       // write ping-pong buffer
    __syncthreads();                   // single barrier per K-step
  }

  // Epilogue: C/D layout VGPR i -> M = i + 8*(lane>=16), N = lane&15.
  const int laneN = lane & 15;
  const int laneH = lane >> 4;
#pragma unroll
  for (int nt = 0; nt < 2; ++nt) {
    int n = blockIdx.y * BN + waveN + nt * 16 + laneN;
    float s = scale[n];
    float b = bias[n];
#pragma unroll
    for (int mt = 0; mt < 4; ++mt) {
      int m0 = blockIdx.x * BM + waveM + mt * 16 + laneH * 8;
#pragma unroll
      for (int i = 0; i < 8; ++i) {
        out[(size_t)(m0 + i) * OUT_F + n] = acc[mt][nt][i] * s + b;
      }
    }
  }
}

extern "C" void kernel_launch(void* const* d_in, const int* in_sizes, int n_in,
                              void* d_out, int out_size, void* d_ws, size_t ws_size,
                              hipStream_t stream) {
  (void)in_sizes; (void)n_in; (void)d_ws; (void)ws_size; (void)out_size;
  const float* input = (const float*)d_in[0];
  const int*   qw    = (const int*)d_in[1];
  const float* sc    = (const float*)d_in[2];
  const float* bi    = (const float*)d_in[3];
  float* out = (float*)d_out;

  // grid.x = M blocks (fast-moving): 64 consecutive blocks share one 2MB W
  // panel; A (134MB) becomes L2-resident (192MB L2) across N panels.
  dim3 grid(TOKENS_N / BM, OUT_F / BN);  // 64 x 86
  bitnet_wmma_gemm<<<grid, 256, 0, stream>>>(input, qw, sc, bi, out);
}
